// MaxSimilarityMatcher_12833362280942
// MI455X (gfx1250) — matrix operand order repaired and checked by reading, not compiled, not run on hardware
//
#include <hip/hip_runtime.h>

typedef _Float16 h16;
typedef __attribute__((ext_vector_type(8)))  _Float16 v8h;
typedef __attribute__((ext_vector_type(16))) _Float16 v16h;
typedef __attribute__((ext_vector_type(8)))  float    v8f;
typedef __attribute__((ext_vector_type(4)))  float    v4f;

#define BATCH 2
#define NA 4096
#define MB 4096
#define CD 128
#define NROT 4
#define INV_TEMP 20.0f
#define THRESH 0.01f
#define EPSN 1e-12f
#define NEG_BIG -3.0e38f
#define NSPLIT 16

// ---------------------------------------------------------------------------
// Pass 0a: L2-normalize f32 rows [nrows][128] -> f16 rows. One wave per row.
// ---------------------------------------------------------------------------
__global__ __launch_bounds__(256) void k_norm_f16(const float* __restrict__ in,
                                                  h16* __restrict__ out, int nrows) {
  int row  = blockIdx.x * (blockDim.x >> 5) + (threadIdx.x >> 5);
  int lane = threadIdx.x & 31;
  if (row >= nrows) return;
  const float* r = in + (size_t)row * CD;
  v4f x = *(const v4f*)(r + lane * 4);
  float ss = x.x * x.x + x.y * x.y + x.z * x.z + x.w * x.w;
#pragma unroll
  for (int off = 16; off > 0; off >>= 1) ss += __shfl_xor(ss, off, 32);
  float inv = rsqrtf(ss + EPSN);
  h16* o = out + (size_t)row * CD + lane * 4;
  o[0] = (h16)(x.x * inv); o[1] = (h16)(x.y * inv);
  o[2] = (h16)(x.z * inv); o[3] = (h16)(x.w * inv);
}

// ---------------------------------------------------------------------------
// Pass 0b: steering  Y[n][d] = sum_c X[n][c] * W[d][c]   (f32, tiny GEMM)
// One row per 128-thread block; X row staged in LDS.
// ---------------------------------------------------------------------------
__global__ __launch_bounds__(128) void k_steer(const float* __restrict__ X,
                                               const float* __restrict__ W,
                                               float* __restrict__ Y) {
  __shared__ float xs[CD];
  int n = blockIdx.x;
  int t = threadIdx.x;
  xs[t] = X[(size_t)n * CD + t];
  __syncthreads();
  const float* wr = W + (size_t)t * CD;
  float acc = 0.f;
#pragma unroll 8
  for (int c = 0; c < CD; ++c) acc += xs[c] * wr[c];
  Y[(size_t)n * CD + t] = acc;
}

// ---------------------------------------------------------------------------
// Pass 1: corr[b,n,m] = INV_TEMP * max_r ( dAn_r[b,n,:] . dBn[b,m,:] )
// Block = 256 threads (8 waves), 128x128 output tile, wave tile = 32x64
// (2x4 WMMA 16x16 tiles). K=128 in 4 steps of 32, f16 WMMA, f32 accum.
// ---------------------------------------------------------------------------
__global__ __launch_bounds__(256) void k_corr(const h16* __restrict__ dAh,   // [4][B][N][C]
                                              const h16* __restrict__ dBh,   // [B][M][C]
                                              float* __restrict__ corr) {    // [B][N][M]
  int b     = blockIdx.z;
  int nTile = blockIdx.y;
  int mTile = blockIdx.x;
  int wave  = threadIdx.x >> 5;
  int lane  = threadIdx.x & 31;
  int wr = wave >> 1;            // 0..3 : 32-row group
  int wc = wave & 1;             // 0..1 : 64-col group
  int rowBase = nTile * 128 + wr * 32;
  int colBase = mTile * 128 + wc * 64;
  int lh  = lane & 15;
  int sel = lane >> 4;           // 0 | 1

  v8f cmax[2][4];
#pragma unroll
  for (int i = 0; i < 2; ++i)
#pragma unroll
    for (int j = 0; j < 4; ++j)
#pragma unroll
      for (int v = 0; v < 8; ++v) cmax[i][j][v] = NEG_BIG;

  const h16* Bb = dBh + (size_t)b * MB * CD;
  for (int rot = 0; rot < NROT; ++rot) {
    const h16* Ab = dAh + (((size_t)rot * BATCH + b) * NA) * CD;
    v8f acc[2][4];
#pragma unroll
    for (int i = 0; i < 2; ++i)
#pragma unroll
      for (int j = 0; j < 4; ++j)
#pragma unroll
        for (int v = 0; v < 8; ++v) acc[i][j][v] = 0.f;

#pragma unroll
    for (int k0 = 0; k0 < CD; k0 += 32) {
      v16h a[2], bf[4];
      // A 16x32 f16 layout: lanes 0-15 hold K {0..7,16..23}; lanes 16-31 {8..15,24..31}
#pragma unroll
      for (int i = 0; i < 2; ++i) {
        const h16* p = Ab + (size_t)(rowBase + i * 16 + lh) * CD + k0 + sel * 8;
        v8h lo = *(const v8h*)p;
        v8h hi = *(const v8h*)(p + 16);
#pragma unroll
        for (int t = 0; t < 8; ++t) { a[i][t] = lo[t]; a[i][t + 8] = hi[t]; }
      }
      // B 32x16 f16 layout: lanes 0-15 hold K {0..7,16..23}; lanes 16-31 {8..15,24..31}
#pragma unroll
      for (int j = 0; j < 4; ++j) {
        const h16* p = Bb + (size_t)(colBase + j * 16 + lh) * CD + k0 + sel * 8;
        v8h lo = *(const v8h*)p;
        v8h hi = *(const v8h*)(p + 16);
#pragma unroll
        for (int t = 0; t < 8; ++t) { bf[j][t] = lo[t]; bf[j][t + 8] = hi[t]; }
      }
#pragma unroll
      for (int i = 0; i < 2; ++i)
#pragma unroll
        for (int j = 0; j < 4; ++j)
          acc[i][j] = __builtin_amdgcn_wmma_f32_16x16x32_f16(
              false, a[i], false, bf[j], (short)0, acc[i][j], false, false);
    }
#pragma unroll
    for (int i = 0; i < 2; ++i)
#pragma unroll
      for (int j = 0; j < 4; ++j)
#pragma unroll
        for (int v = 0; v < 8; ++v)
          cmax[i][j][v] = fmaxf(cmax[i][j][v], acc[i][j][v]);
  }

  // C/D layout: VGPR v -> row (v + 8*sel), col = lane&15
  float* out = corr + (size_t)b * NA * MB;
#pragma unroll
  for (int i = 0; i < 2; ++i)
#pragma unroll
    for (int j = 0; j < 4; ++j)
#pragma unroll
      for (int v = 0; v < 8; ++v) {
        int rr = rowBase + i * 16 + v + sel * 8;
        int cc = colBase + j * 16 + lh;
        out[(size_t)rr * MB + cc] = INV_TEMP * cmax[i][j][v];
      }
}

// ---------------------------------------------------------------------------
// Pass 2: per-row (axis=-1) online softmax stats. One wave per row.
// ---------------------------------------------------------------------------
__global__ __launch_bounds__(256) void k_rowstats(const float* __restrict__ corr,
                                                  float* __restrict__ rowmax,
                                                  float* __restrict__ rowsum) {
  int row  = blockIdx.x * (blockDim.x >> 5) + (threadIdx.x >> 5);
  int lane = threadIdx.x & 31;
  if (row >= BATCH * NA) return;
  const float* r = corr + (size_t)row * MB;
  float rm = NEG_BIG, rs = 0.f;
  for (int i = 0; i < MB / 128; ++i) {
    v4f x = *(const v4f*)(r + (i * 32 + lane) * 4);
#pragma unroll
    for (int t = 0; t < 4; ++t) {
      float v = x[t];
      if (v > rm) { rs = rs * __expf(rm - v) + 1.f; rm = v; }
      else        { rs += __expf(v - rm); }
    }
  }
#pragma unroll
  for (int off = 16; off > 0; off >>= 1) {
    float om = __shfl_xor(rm, off, 32);
    float os = __shfl_xor(rs, off, 32);
    float nm = fmaxf(rm, om);
    rs = rs * __expf(rm - nm) + os * __expf(om - nm);
    rm = nm;
  }
  if (lane == 0) { rowmax[row] = rm; rowsum[row] = rs; }
}

// ---------------------------------------------------------------------------
// Pass 3: per-column (axis=-2) stats via NSPLIT partials, then combine.
// ---------------------------------------------------------------------------
__global__ __launch_bounds__(256) void k_colpart_ms(const float* __restrict__ corr,
                                                    float* __restrict__ pmax,
                                                    float* __restrict__ psum) {
  int col = blockIdx.x * blockDim.x + threadIdx.x;   // b*MB + m
  if (col >= BATCH * MB) return;
  int b = col >> 12, m = col & (MB - 1);
  int split = blockIdx.y;
  const float* p = corr + ((size_t)b * NA + (size_t)split * (NA / NSPLIT)) * MB + m;
  float rm = NEG_BIG, rs = 0.f;
  for (int n = 0; n < NA / NSPLIT; ++n) {
    float v = p[(size_t)n * MB];
    if (v > rm) { rs = rs * __expf(rm - v) + 1.f; rm = v; }
    else        { rs += __expf(v - rm); }
  }
  pmax[(size_t)split * BATCH * MB + col] = rm;
  psum[(size_t)split * BATCH * MB + col] = rs;
}

__global__ __launch_bounds__(256) void k_colcomb_ms(const float* __restrict__ pmax,
                                                    const float* __restrict__ psum,
                                                    float* __restrict__ colmax,
                                                    float* __restrict__ colsum) {
  int col = blockIdx.x * blockDim.x + threadIdx.x;
  if (col >= BATCH * MB) return;
  float rm = NEG_BIG, rs = 0.f;
#pragma unroll
  for (int s = 0; s < NSPLIT; ++s) {
    float om = pmax[(size_t)s * BATCH * MB + col];
    float os = psum[(size_t)s * BATCH * MB + col];
    float nm = fmaxf(rm, om);
    rs = rs * __expf(rm - nm) + os * __expf(om - nm);
    rm = nm;
  }
  colmax[col] = rm; colsum[col] = rs;
}

// ---------------------------------------------------------------------------
// Pass 4: in-place dual softmax  P = exp(2c - rm - cm) / (rs*cs)
// ---------------------------------------------------------------------------
__global__ __launch_bounds__(256) void k_dualsoftmax(float* __restrict__ P,
                                                     const float* __restrict__ rowmax,
                                                     const float* __restrict__ rowsum,
                                                     const float* __restrict__ colmax,
                                                     const float* __restrict__ colsum) {
  size_t idx = ((size_t)blockIdx.x * blockDim.x + threadIdx.x) * 4;
  size_t rown = idx >> 12;          // b*NA + n
  int m = (int)(idx & (MB - 1));
  int b = (int)(rown >> 12);
  v4f c = *(const v4f*)(P + idx);
  float rm = rowmax[rown], rs = rowsum[rown];
  v4f o;
#pragma unroll
  for (int t = 0; t < 4; ++t) {
    size_t ci = (size_t)b * MB + (m + t);
    o[t] = __expf(2.f * c[t] - rm - colmax[ci]) / (rs * colsum[ci]);
  }
  *(v4f*)(P + idx) = o;
}

// ---------------------------------------------------------------------------
// Pass 5a: per-row max + argmax of P (one wave per row, first-index ties)
// ---------------------------------------------------------------------------
__global__ __launch_bounds__(256) void k_prowstats(const float* __restrict__ P,
                                                   float* __restrict__ prmax,
                                                   int* __restrict__ pargr) {
  int row  = blockIdx.x * (blockDim.x >> 5) + (threadIdx.x >> 5);
  int lane = threadIdx.x & 31;
  if (row >= BATCH * NA) return;
  const float* r = P + (size_t)row * MB;
  float bv = -1.f; int bi = 0;
  for (int i = 0; i < MB / 128; ++i) {
    int base = (i * 32 + lane) * 4;
    v4f x = *(const v4f*)(r + base);
#pragma unroll
    for (int t = 0; t < 4; ++t)
      if (x[t] > bv) { bv = x[t]; bi = base + t; }
  }
#pragma unroll
  for (int off = 16; off > 0; off >>= 1) {
    float ov = __shfl_xor(bv, off, 32);
    int   oi = __shfl_xor(bi, off, 32);
    if (ov > bv || (ov == bv && oi < bi)) { bv = ov; bi = oi; }
  }
  if (lane == 0) { prmax[row] = bv; pargr[row] = bi; }
}

// ---------------------------------------------------------------------------
// Pass 5b: per-column max of P via partials
// ---------------------------------------------------------------------------
__global__ __launch_bounds__(256) void k_colpart_max(const float* __restrict__ P,
                                                     float* __restrict__ pmax) {
  int col = blockIdx.x * blockDim.x + threadIdx.x;
  if (col >= BATCH * MB) return;
  int b = col >> 12, m = col & (MB - 1);
  int split = blockIdx.y;
  const float* p = P + ((size_t)b * NA + (size_t)split * (NA / NSPLIT)) * MB + m;
  float rm = -1.f;
  for (int n = 0; n < NA / NSPLIT; ++n) rm = fmaxf(rm, p[(size_t)n * MB]);
  pmax[(size_t)split * BATCH * MB + col] = rm;
}

__global__ __launch_bounds__(256) void k_colcomb_max(const float* __restrict__ pmax,
                                                     float* __restrict__ pcmax) {
  int col = blockIdx.x * blockDim.x + threadIdx.x;
  if (col >= BATCH * MB) return;
  float rm = -1.f;
#pragma unroll
  for (int s = 0; s < NSPLIT; ++s) rm = fmaxf(rm, pmax[(size_t)s * BATCH * MB + col]);
  pcmax[col] = rm;
}

// ---------------------------------------------------------------------------
// Pass 6: mutual mask
// ---------------------------------------------------------------------------
__global__ __launch_bounds__(256) void k_mutual(const float* __restrict__ P,
                                                const float* __restrict__ prmax,
                                                const float* __restrict__ pcmax,
                                                float* __restrict__ mut) {
  size_t idx = ((size_t)blockIdx.x * blockDim.x + threadIdx.x) * 4;
  size_t rown = idx >> 12;
  int m = (int)(idx & (MB - 1));
  int b = (int)(rown >> 12);
  v4f p = *(const v4f*)(P + idx);
  float rm = prmax[rown];
  v4f o;
#pragma unroll
  for (int t = 0; t < 4; ++t) {
    float cm = pcmax[(size_t)b * MB + (m + t)];
    o[t] = (p[t] == rm && p[t] == cm && p[t] > THRESH) ? 1.f : 0.f;
  }
  *(v4f*)(mut + idx) = o;
}

// ---------------------------------------------------------------------------
// Pass 7: keypoints_A passthrough + matches_B gather
// ---------------------------------------------------------------------------
__global__ __launch_bounds__(256) void k_matches(const float* __restrict__ kA,
                                                 const float* __restrict__ kB,
                                                 const int* __restrict__ pargr,
                                                 float* __restrict__ outKA,
                                                 float* __restrict__ outMB) {
  int i = blockIdx.x * blockDim.x + threadIdx.x;    // b*NA + n
  if (i >= BATCH * NA) return;
  outKA[2 * i]     = kA[2 * i];
  outKA[2 * i + 1] = kA[2 * i + 1];
  int b = i >> 12;
  size_t src = ((size_t)b * MB + pargr[i]) * 2;
  outMB[2 * i]     = kB[src];
  outMB[2 * i + 1] = kB[src + 1];
}

// ---------------------------------------------------------------------------
extern "C" void kernel_launch(void* const* d_in, const int* in_sizes, int n_in,
                              void* d_out, int out_size, void* d_ws, size_t ws_size,
                              hipStream_t stream) {
  const float* kA = (const float*)d_in[0];
  const float* dA = (const float*)d_in[1];
  const float* kB = (const float*)d_in[2];
  const float* dB = (const float*)d_in[3];
  const float* W  = (const float*)d_in[4];

  const size_t PN = (size_t)BATCH * NA * MB;
  float* P     = (float*)d_out;          // [B][N][M]
  float* mut   = P + PN;                 // [B][N][M]  (also big scratch until pass 6)
  float* outKA = P + 2 * PN;             // [B][N][2]
  float* outMB = outKA + (size_t)BATCH * NA * 2;

  // ---- big scratch lives in the mutual half of d_out (consumed before pass 6)
  char* scr = (char*)mut;
  h16*   dBh    = (h16*)scr;                               // 2 MB  [B][M][C] f16
  h16*   dAh    = (h16*)(scr + (size_t)(2u  << 20));       // 8 MB  [4][B][N][C] f16
  float* Xb     = (float*)(scr + (size_t)(10u << 20));     // 12 MB X1..X3 f32
  float* rowmax = (float*)(scr + (size_t)(22u << 20));
  float* rowsum = rowmax + BATCH * NA;
  float* colmax = rowsum + BATCH * NA;
  float* colsum = colmax + BATCH * MB;
  float* pmaxp  = colsum + BATCH * MB;                     // NSPLIT * B*MB
  float* psump  = pmaxp + (size_t)NSPLIT * BATCH * MB;

  // ---- tiny scratch read while mutual is written lives in d_ws (96 KB)
  float* prmax = (float*)d_ws;
  int*   pargr = (int*)(prmax + BATCH * NA);
  float* pcmax = (float*)(pargr + BATCH * NA);

  const size_t XR  = (size_t)BATCH * NA * CD;   // floats per rot buffer
  const size_t RST = (size_t)BATCH * NA * CD;   // halfs per rot in dAh

  // Pass 0: normalize B, build steered+normalized A (4 rots)
  k_norm_f16<<<(BATCH * MB) / 8, 256, 0, stream>>>(dB, dBh, BATCH * MB);
  k_norm_f16<<<(BATCH * NA) / 8, 256, 0, stream>>>(dA, dAh, BATCH * NA);
  k_steer<<<BATCH * NA, CD, 0, stream>>>(dA, W, Xb);                       // X1
  k_norm_f16<<<(BATCH * NA) / 8, 256, 0, stream>>>(Xb, dAh + RST, BATCH * NA);
  k_steer<<<BATCH * NA, CD, 0, stream>>>(Xb, W, Xb + XR);                  // X2
  k_norm_f16<<<(BATCH * NA) / 8, 256, 0, stream>>>(Xb + XR, dAh + 2 * RST, BATCH * NA);
  k_steer<<<BATCH * NA, CD, 0, stream>>>(Xb + XR, W, Xb + 2 * XR);         // X3
  k_norm_f16<<<(BATCH * NA) / 8, 256, 0, stream>>>(Xb + 2 * XR, dAh + 3 * RST, BATCH * NA);

  // Pass 1: WMMA correlation with rot-max (writes corr into P region)
  k_corr<<<dim3(MB / 128, NA / 128, BATCH), 256, 0, stream>>>(dAh, dBh, P);

  // Pass 2-3: softmax stats
  k_rowstats<<<(BATCH * NA) / 8, 256, 0, stream>>>(P, rowmax, rowsum);
  k_colpart_ms<<<dim3((BATCH * MB) / 256, NSPLIT), 256, 0, stream>>>(P, pmaxp, psump);
  k_colcomb_ms<<<(BATCH * MB) / 256, 256, 0, stream>>>(pmaxp, psump, colmax, colsum);

  // Pass 4: dual softmax in place
  k_dualsoftmax<<<(unsigned)(PN / 1024), 256, 0, stream>>>(P, rowmax, rowsum, colmax, colsum);

  // Pass 5: P row max/argmax + col max
  k_prowstats<<<(BATCH * NA) / 8, 256, 0, stream>>>(P, prmax, pargr);
  k_colpart_max<<<dim3((BATCH * MB) / 256, NSPLIT), 256, 0, stream>>>(P, pmaxp);
  k_colcomb_max<<<(BATCH * MB) / 256, 256, 0, stream>>>(pmaxp, pcmax);

  // Pass 6: mutual mask (overwrites the scratch half)
  k_mutual<<<(unsigned)(PN / 1024), 256, 0, stream>>>(P, prmax, pcmax, mut);

  // Pass 7: keypoint outputs
  k_matches<<<(BATCH * NA) / 256, 256, 0, stream>>>(kA, kB, pargr, outKA, outMB);
}
